// GATLayer_15556371546272
// MI455X (gfx1250) — matrix operand order repaired and checked by reading, not compiled, not run on hardware
//
#include <hip/hip_runtime.h>
#include <hip/hip_bf16.h>
#include <math.h>

typedef __attribute__((ext_vector_type(16))) __bf16 v16bf;
typedef __attribute__((ext_vector_type(8)))  float  v8f;

#define IN_F   128     // input feature dim (= NUM_HEADS*OUT_FEATS too)
#define NHEAD  4
#define DHEAD  32
#define NEG_SLOPE 0.2f

// ---------------------------------------------------------------------------
// Kernel 0: pre-split W into bf16 hi/lo, stored in WMMA B-operand layout.
// combo = kc*8+nt (32 combos). Per lane: 16 bf16 values = the exact v16bf
// register image for that (kc, nt, lane). 1024 threads total, runs once.
// B k order used here (same as A, 32x16, 16-bit): elem e of lane L -> K = (L>>4)*16+e, N = L&15.
// ---------------------------------------------------------------------------
__global__ __launch_bounds__(256) void gat_prepw(const float* __restrict__ W,
                                                 __bf16* __restrict__ Whi,
                                                 __bf16* __restrict__ Wlo) {
    const int tid   = blockIdx.x * blockDim.x + threadIdx.x;  // 0..1023
    const int lane  = tid & 31;
    const int combo = tid >> 5;                               // kc*8 + nt
    if (combo >= 32) return;
    const int kc = combo >> 3;
    const int nt = combo & 7;
    const float* brow = W + (size_t)(nt * 16 + (lane & 15)) * IN_F
                          + kc * 32 + (lane >> 4) * 16;
    __bf16* hp = Whi + ((size_t)combo * 32 + lane) * 16;
    __bf16* lp = Wlo + ((size_t)combo * 32 + lane) * 16;
#pragma unroll
    for (int e = 0; e < 16; ++e) {
        const float x  = brow[e];
        const __bf16 hi = (__bf16)x;
        hp[e] = hi;
        lp[e] = (__bf16)(x - (float)hi);
    }
}

// ---------------------------------------------------------------------------
// Kernel 1: h = feat @ W^T via bf16x3 WMMA (f32-accurate split multiplication)
// One wave handles a 16-node (M) tile across all 128 output cols (8 N-tiles).
// B operands come pre-split/pre-laid-out from workspace (2 b128 loads each).
// ---------------------------------------------------------------------------
__global__ __launch_bounds__(256) void gat_gemm(const float* __restrict__ feat,
                                                const __bf16* __restrict__ Whi,
                                                const __bf16* __restrict__ Wlo,
                                                float* __restrict__ h,
                                                int nTiles) {
    const int wave = (blockIdx.x * blockDim.x + threadIdx.x) >> 5;
    const int lane = threadIdx.x & 31;
    if (wave >= nTiles) return;                 // wave-uniform: EXEC stays all-1s

    const int m    = lane & 15;                 // row within tile (A) / col (C)
    const int half = lane >> 4;                 // 0: lanes 0-15, 1: lanes 16-31
    const int node = wave * 16 + m;

    v8f acc[8];
#pragma unroll
    for (int nt = 0; nt < 8; ++nt) acc[nt] = (v8f){0.f,0.f,0.f,0.f,0.f,0.f,0.f,0.f};

#pragma unroll
    for (int kc = 0; kc < 4; ++kc) {            // K chunks of 32
        // ---- A tile (16x32 bf16 layout): lane half selects K subgroup ----
        const float* arow = feat + (size_t)node * IN_F + kc * 32 + half * 16;
        float av[16];
#pragma unroll
        for (int j = 0; j < 8; ++j) {           // VGPR j holds K = off, off+1
            const int off = 2 * j;
            av[2 * j]     = arow[off];
            av[2 * j + 1] = arow[off + 1];
        }
        v16bf ahi, alo;
#pragma unroll
        for (int e = 0; e < 16; ++e) {
            const float x = av[e];
            const __bf16 hi = (__bf16)x;
            ahi[e] = hi;
            alo[e] = (__bf16)(x - (float)hi);
        }

#pragma unroll
        for (int nt = 0; nt < 8; ++nt) {
            const size_t boff = ((size_t)(kc * 8 + nt) * 32 + lane) * 16;
            const v16bf bhi = *(const v16bf*)(Whi + boff);
            const v16bf blo = *(const v16bf*)(Wlo + boff);
            // bf16x3: AhiBhi + AloBhi + AhiBlo  (f32 accumulate)
            acc[nt] = __builtin_amdgcn_wmma_f32_16x16x32_bf16(
                false, ahi, false, bhi, (short)0, acc[nt], false, false);
            acc[nt] = __builtin_amdgcn_wmma_f32_16x16x32_bf16(
                false, alo, false, bhi, (short)0, acc[nt], false, false);
            acc[nt] = __builtin_amdgcn_wmma_f32_16x16x32_bf16(
                false, ahi, false, blo, (short)0, acc[nt], false, false);
        }
    }

    // ---- store C: VGPR r -> row (r + half*8), col = nt*16 + (lane&15) ----
#pragma unroll
    for (int nt = 0; nt < 8; ++nt) {
#pragma unroll
        for (int r = 0; r < 8; ++r) {
            const int nrow = wave * 16 + r + half * 8;
            h[(size_t)nrow * IN_F + nt * 16 + m] = acc[nt][r];
        }
    }
}

// ---------------------------------------------------------------------------
// Kernel 2: per-node logits el/er, init emax/denom, zero output
// One wave per node; lane = feature d within a head.
// ---------------------------------------------------------------------------
__global__ __launch_bounds__(256) void gat_node(const float* __restrict__ h,
                                                const float* __restrict__ attn_l,
                                                const float* __restrict__ attn_r,
                                                float* __restrict__ el,
                                                float* __restrict__ er,
                                                float* __restrict__ emax,
                                                float* __restrict__ denom,
                                                float* __restrict__ out,
                                                int nNodes) {
    const int node = (blockIdx.x * blockDim.x + threadIdx.x) >> 5;
    const int lane = threadIdx.x & 31;
    if (node >= nNodes) return;

    float lacc[NHEAD], racc[NHEAD];
#pragma unroll
    for (int hd = 0; hd < NHEAD; ++hd) {
        const float hv = h[(size_t)node * IN_F + hd * DHEAD + lane];
        lacc[hd] = hv * attn_l[hd * DHEAD + lane];
        racc[hd] = hv * attn_r[hd * DHEAD + lane];
        out[(size_t)node * IN_F + hd * DHEAD + lane] = 0.f;
    }
#pragma unroll
    for (int hd = 0; hd < NHEAD; ++hd) {
        float lv = lacc[hd], rv = racc[hd];
#pragma unroll
        for (int off = 16; off > 0; off >>= 1) {
            lv += __shfl_xor(lv, off, 32);
            rv += __shfl_xor(rv, off, 32);
        }
        if (lane == 0) { el[node * NHEAD + hd] = lv; er[node * NHEAD + hd] = rv; }
    }
    if (lane < NHEAD) {
        emax[node * NHEAD + lane]  = -__builtin_inff();
        denom[node * NHEAD + lane] = 0.f;
    }
}

// float atomic max via signed/unsigned int ordering trick (works for +/-)
__device__ __forceinline__ void atomicMaxF(float* addr, float v) {
    if (v >= 0.f) atomicMax((int*)addr, __float_as_int(v));
    else          atomicMin((unsigned int*)addr, __float_as_uint(v));
}

// ---------------------------------------------------------------------------
// Kernel 3: per-edge leaky-relu logits -> segment max over dst
// ---------------------------------------------------------------------------
__global__ __launch_bounds__(256) void gat_emax(const int* __restrict__ src,
                                                const int* __restrict__ dst,
                                                const float* __restrict__ el,
                                                const float* __restrict__ er,
                                                float* __restrict__ emax,
                                                int E) {
    const int e = blockIdx.x * blockDim.x + threadIdx.x;
    if (e >= E) return;
    const int s = src[e], t = dst[e];
    const float4 ls = ((const float4*)el)[s];
    const float4 rt = ((const float4*)er)[t];
    float v[4] = {ls.x + rt.x, ls.y + rt.y, ls.z + rt.z, ls.w + rt.w};
#pragma unroll
    for (int hd = 0; hd < NHEAD; ++hd) {
        float x = v[hd];
        x = (x > 0.f) ? x : NEG_SLOPE * x;
        atomicMaxF(&emax[t * NHEAD + hd], x);
    }
}

// ---------------------------------------------------------------------------
// Kernel 4: denom = segment_sum(exp(e - emax[dst]))
// ---------------------------------------------------------------------------
__global__ __launch_bounds__(256) void gat_denom(const int* __restrict__ src,
                                                 const int* __restrict__ dst,
                                                 const float* __restrict__ el,
                                                 const float* __restrict__ er,
                                                 const float* __restrict__ emax,
                                                 float* __restrict__ denom,
                                                 int E) {
    const int e = blockIdx.x * blockDim.x + threadIdx.x;
    if (e >= E) return;
    const int s = src[e], t = dst[e];
    const float4 ls = ((const float4*)el)[s];
    const float4 rt = ((const float4*)er)[t];
    const float4 mx = ((const float4*)emax)[t];
    float v[4]  = {ls.x + rt.x, ls.y + rt.y, ls.z + rt.z, ls.w + rt.w};
    float mv[4] = {mx.x, mx.y, mx.z, mx.w};
#pragma unroll
    for (int hd = 0; hd < NHEAD; ++hd) {
        float x = v[hd];
        x = (x > 0.f) ? x : NEG_SLOPE * x;
        atomicAdd(&denom[t * NHEAD + hd], expf(x - mv[hd]));
    }
}

// ---------------------------------------------------------------------------
// Kernel 5: h_out[dst] += alpha * h[src]   (wave per edge, lane = feature d)
// ---------------------------------------------------------------------------
__global__ __launch_bounds__(256) void gat_agg(const int* __restrict__ src,
                                               const int* __restrict__ dst,
                                               const float* __restrict__ el,
                                               const float* __restrict__ er,
                                               const float* __restrict__ emax,
                                               const float* __restrict__ denom,
                                               const float* __restrict__ h,
                                               float* __restrict__ out,
                                               int E) {
    const int e    = (blockIdx.x * blockDim.x + threadIdx.x) >> 5;
    const int lane = threadIdx.x & 31;
    if (e >= E) return;
    const int s = src[e], t = dst[e];

    float alpha = 0.f;
    if (lane < NHEAD) {
        float x = el[s * NHEAD + lane] + er[t * NHEAD + lane];
        x = (x > 0.f) ? x : NEG_SLOPE * x;
        alpha = expf(x - emax[t * NHEAD + lane]) / denom[t * NHEAD + lane];
    }
    float a[NHEAD];
#pragma unroll
    for (int hd = 0; hd < NHEAD; ++hd) a[hd] = __shfl(alpha, hd, 32);

#pragma unroll
    for (int hd = 0; hd < NHEAD; ++hd) {
        const float hv = h[(size_t)s * IN_F + hd * DHEAD + lane];
        atomicAdd(&out[(size_t)t * IN_F + hd * DHEAD + lane], a[hd] * hv);
    }
}

// ---------------------------------------------------------------------------
extern "C" void kernel_launch(void* const* d_in, const int* in_sizes, int n_in,
                              void* d_out, int out_size, void* d_ws, size_t ws_size,
                              hipStream_t stream) {
    const float* feat   = (const float*)d_in[0];
    const int*   src    = (const int*)  d_in[1];
    const int*   dst    = (const int*)  d_in[2];
    const float* W      = (const float*)d_in[3];
    const float* attn_l = (const float*)d_in[4];
    const float* attn_r = (const float*)d_in[5];
    float*       out    = (float*)d_out;

    const int N = in_sizes[0] / IN_F;   // 100000 nodes
    const int E = in_sizes[1];          // 1600000 edges

    // workspace layout: [Whi 32KB][Wlo 32KB][h N*128 f32][el|er|emax|denom N*4]
    __bf16* Whi = (__bf16*)d_ws;                         // 128*128 bf16
    __bf16* Wlo = Whi + (size_t)IN_F * IN_F;             // 128*128 bf16
    float*  h     = (float*)(Wlo + (size_t)IN_F * IN_F); // 32B-aligned (64KB off)
    float*  el    = h    + (size_t)N * IN_F;
    float*  er    = el   + (size_t)N * NHEAD;
    float*  emax  = er   + (size_t)N * NHEAD;
    float*  denom = emax + (size_t)N * NHEAD;

    const int nTiles = (N + 15) / 16;   // 6250 (exact)

    // 0) one-time W bf16 hi/lo split into WMMA B layout (1024 threads)
    gat_prepw<<<4, 256, 0, stream>>>(W, Whi, Wlo);
    // 1) projection GEMM (WMMA bf16x3), 8 waves / block
    gat_gemm<<<dim3((nTiles + 7) / 8), 256, 0, stream>>>(feat, Whi, Wlo, h, nTiles);
    // 2) node logits + init
    gat_node<<<dim3((N + 7) / 8), 256, 0, stream>>>(h, attn_l, attn_r, el, er, emax, denom, out, N);
    // 3) segment max
    gat_emax<<<dim3((E + 255) / 256), 256, 0, stream>>>(src, dst, el, er, emax, E);
    // 4) segment sum of exp
    gat_denom<<<dim3((E + 255) / 256), 256, 0, stream>>>(src, dst, el, er, emax, denom, E);
    // 5) weighted aggregation (wave per edge)
    gat_agg<<<dim3((E + 7) / 8), 256, 0, stream>>>(src, dst, el, er, emax, denom, h, out, E);
}
